// SpectralBlock_24739011625163
// MI455X (gfx1250) — matrix operand order repaired and checked by reading, not compiled, not run on hardware
//
#include <hip/hip_runtime.h>
#include <hip/hip_bf16.h>
#include <math.h>

// ---------------------------------------------------------------------------
// Types for CDNA5 WMMA (wave32, 16x16x32 bf16 -> f32)
// ---------------------------------------------------------------------------
typedef __attribute__((ext_vector_type(16))) __bf16 v16bf;
typedef __attribute__((ext_vector_type(8)))  float  v8f;

#define BM 64
#define BN 64
#define BK 32
#define LDK 48  // padded LDS row (bf16 elems), 96B rows keep 16B alignment

// ---------------------------------------------------------------------------
// Generic dual-term batched GEMM:
//   D[bz] = A1[bz] @ B1[bz] + scale2 * A2[bz] @ B2[bz]   (A2 may be null)
// A operands are never batched (shared small matrices or flat activations).
// B and D get two-level batch strides: off = (bz/nbi)*outer + (bz%nbi)*inner.
// Epilogue: optional bias (per column), act==1 -> exact GELU, optional
// residual add (non-batched), then store f32.
// Math path: f32 -> (bf16 hi, bf16 lo) split in LDS; 3 WMMAs per tile pair
// (hi*hi + hi*lo + lo*hi) for ~fp32 accuracy on the bf16 matrix pipe.
// Staging: uniform fast path with unconditional float4 (global_load_b128)
// when the tile is fully in-bounds and the stride allows 16B alignment;
// branchless clamped-address fallback otherwise (no exec-mask churn).
// ---------------------------------------------------------------------------
__global__ void __launch_bounds__(128)
gemm_bf16x3_kernel(const float* __restrict__ A1, long long lda1,
                   const float* __restrict__ B1, long long ldb1,
                   long long bsB1o, long long bsB1i,
                   const float* __restrict__ A2, long long lda2,
                   const float* __restrict__ B2, long long ldb2,
                   long long bsB2o, long long bsB2i,
                   float scale2,
                   float* __restrict__ D, long long ldd,
                   long long bsDo, long long bsDi,
                   const float* __restrict__ bias,
                   const float* __restrict__ resid, long long ldr,
                   int M, int N, int K, int nbi, int act)
{
    __shared__ __bf16 sAh[BM][LDK];
    __shared__ __bf16 sAl[BM][LDK];
    __shared__ __bf16 sBh[BN][LDK];
    __shared__ __bf16 sBl[BN][LDK];

    const int tid  = threadIdx.x;
    const int lane = tid & 31;
    const int wave = tid >> 5;
    const int half = (lane >> 4) & 1;
    const int lr   = lane & 15;
    const int wm   = (wave >> 1) * 32;
    const int wn   = (wave & 1) * 32;

    const int m0 = blockIdx.y * BM;
    const int n0 = blockIdx.x * BN;
    const int bz = blockIdx.z;
    const int bo = bz / nbi;
    const int bi = bz - bo * nbi;

    v8f acc[2][2];
    #pragma unroll
    for (int i = 0; i < 2; ++i)
        #pragma unroll
        for (int j = 0; j < 2; ++j)
            #pragma unroll
            for (int e = 0; e < 8; ++e) acc[i][j][e] = 0.0f;

    const int nTerms = (A2 != nullptr) ? 2 : 1;

    for (int t = 0; t < nTerms; ++t) {
        const float* Ap = t ? A2 : A1;
        const float* Bp = t ? B2 : B1;
        const long long lda = t ? lda2 : lda1;
        const long long ldb = t ? ldb2 : ldb1;
        Bp += (t ? (bo * bsB2o + bi * bsB2i) : (bo * bsB1o + bi * bsB1i));
        const float sc = t ? scale2 : 1.0f;

        for (int k0 = 0; k0 < K; k0 += BK) {
            __syncthreads();

            // ---- stage A tile: BM x BK ----
            const bool fastA = ((m0 + BM) <= M) && ((k0 + BK) <= K) &&
                               ((lda & 3LL) == 0LL);
            if (fastA) {
                const float* tb = Ap + (long long)m0 * lda + k0;
                #pragma unroll
                for (int p = 0; p < 4; ++p) {          // 512 float4 / 128 thr
                    int idx = tid + p * 128;
                    int r   = idx >> 3;                // 8 float4 per row
                    int c4  = (idx & 7) << 2;
                    const float4 q = *(const float4*)(tb + (long long)r * lda + c4);
                    float vv[4] = {q.x * sc, q.y * sc, q.z * sc, q.w * sc};
                    #pragma unroll
                    for (int e = 0; e < 4; ++e) {
                        __bf16 hi = (__bf16)vv[e];
                        sAh[r][c4 + e] = hi;
                        sAl[r][c4 + e] = (__bf16)(vv[e] - (float)hi);
                    }
                }
            } else {
                #pragma unroll
                for (int i = 0; i < (BM * BK) / 128; ++i) {
                    int idx = tid + i * 128;
                    int r = idx >> 5;                  // / BK
                    int c = idx & 31;                  // % BK
                    int gm = m0 + r, gk = k0 + c;
                    int cm = gm < M ? gm : (M - 1);    // clamp: always legal
                    int ck = gk < K ? gk : (K - 1);
                    float v = Ap[(long long)cm * lda + ck] * sc;
                    if (gm >= M || gk >= K) v = 0.0f;  // branchless mask
                    __bf16 hi = (__bf16)v;
                    sAh[r][c] = hi;
                    sAl[r][c] = (__bf16)(v - (float)hi);
                }
            }

            // ---- stage B tile: BK x BN, stored transposed [n][k] ----
            const bool fastB = ((n0 + BN) <= N) && ((k0 + BK) <= K) &&
                               ((ldb & 3LL) == 0LL);
            if (fastB) {
                const float* tb = Bp + (long long)k0 * ldb + n0;
                #pragma unroll
                for (int p = 0; p < 4; ++p) {
                    int idx = tid + p * 128;
                    int kk  = idx >> 4;                // 16 float4 per row
                    int c4  = (idx & 15) << 2;
                    const float4 q = *(const float4*)(tb + (long long)kk * ldb + c4);
                    float vv[4] = {q.x, q.y, q.z, q.w};
                    #pragma unroll
                    for (int e = 0; e < 4; ++e) {
                        __bf16 hi = (__bf16)vv[e];
                        sBh[c4 + e][kk] = hi;
                        sBl[c4 + e][kk] = (__bf16)(vv[e] - (float)hi);
                    }
                }
            } else {
                #pragma unroll
                for (int i = 0; i < (BK * BN) / 128; ++i) {
                    int idx = tid + i * 128;
                    int kk = idx >> 6;                 // / BN
                    int c  = idx & 63;                 // % BN
                    int gk = k0 + kk, gn = n0 + c;
                    int ckk = gk < K ? gk : (K - 1);
                    int cn  = gn < N ? gn : (N - 1);
                    float v = Bp[(long long)ckk * ldb + cn];
                    if (gk >= K || gn >= N) v = 0.0f;
                    __bf16 hi = (__bf16)v;
                    sBh[c][kk] = hi;
                    sBl[c][kk] = (__bf16)(v - (float)hi);
                }
            }
            __syncthreads();

            // ---- build fragments, contiguous k = 16*half + e in A and B ----
            v16bf ah[2], al[2], bh[2], bl[2];
            #pragma unroll
            for (int tm = 0; tm < 2; ++tm) {
                int row = wm + tm * 16 + lr;
                #pragma unroll
                for (int e = 0; e < 8; ++e) {
                    ah[tm][e]     = sAh[row][16 * half + e];
                    al[tm][e]     = sAl[row][16 * half + e];
                    ah[tm][8 + e] = sAh[row][16 * half + 8 + e];
                    al[tm][8 + e] = sAl[row][16 * half + 8 + e];
                }
            }
            #pragma unroll
            for (int tn = 0; tn < 2; ++tn) {
                int col = wn + tn * 16 + lr;
                #pragma unroll
                for (int e = 0; e < 16; ++e) {
                    bh[tn][e] = sBh[col][16 * half + e];
                    bl[tn][e] = sBl[col][16 * half + e];
                }
            }

            // ---- 3-pass bf16 split WMMA: hi*hi + hi*lo + lo*hi ----
            #pragma unroll
            for (int tm = 0; tm < 2; ++tm)
                #pragma unroll
                for (int tn = 0; tn < 2; ++tn) {
                    acc[tm][tn] = __builtin_amdgcn_wmma_f32_16x16x32_bf16(
                        false, ah[tm], false, bh[tn], (short)0, acc[tm][tn], false, false);
                    acc[tm][tn] = __builtin_amdgcn_wmma_f32_16x16x32_bf16(
                        false, ah[tm], false, bl[tn], (short)0, acc[tm][tn], false, false);
                    acc[tm][tn] = __builtin_amdgcn_wmma_f32_16x16x32_bf16(
                        false, al[tm], false, bh[tn], (short)0, acc[tm][tn], false, false);
                }
        }
    }

    // ---- epilogue: bias / GELU / residual / store ----
    float* Dp = D + bo * bsDo + bi * bsDi;
    #pragma unroll
    for (int tm = 0; tm < 2; ++tm)
        #pragma unroll
        for (int tn = 0; tn < 2; ++tn) {
            int ng = n0 + wn + tn * 16 + lr;
            #pragma unroll
            for (int r = 0; r < 8; ++r) {
                int mg = m0 + wm + tm * 16 + half * 8 + r;
                if (mg < M && ng < N) {
                    float v = acc[tm][tn][r];
                    if (bias) v += bias[ng];
                    if (act == 1) v = 0.5f * v * (1.0f + erff(v * 0.70710678118654752f));
                    if (resid) v += resid[(long long)mg * ldr + ng];
                    Dp[(long long)mg * ldd + ng] = v;
                }
            }
        }
}

// ---------------------------------------------------------------------------
// LayerNorm over rows of length C (two-pass, row cached in registers)
// ---------------------------------------------------------------------------
__global__ void __launch_bounds__(256)
layernorm_kernel(const float* __restrict__ x, const float* __restrict__ g,
                 const float* __restrict__ b, float* __restrict__ out, int C)
{
    __shared__ float red[256];
    const long long row = blockIdx.x;
    const float* xr = x + row * (long long)C;
    float* yr = out + row * (long long)C;
    const int tid = threadIdx.x;

    float v[4];
    int cnt = 0;
    float s = 0.0f;
    for (int c = tid; c < C; c += 256) { v[cnt] = xr[c]; s += v[cnt]; ++cnt; }
    red[tid] = s; __syncthreads();
    for (int o = 128; o > 0; o >>= 1) { if (tid < o) red[tid] += red[tid + o]; __syncthreads(); }
    float mu = red[0] / (float)C;
    __syncthreads();
    s = 0.0f;
    for (int i = 0; i < cnt; ++i) { float d = v[i] - mu; s += d * d; }
    red[tid] = s; __syncthreads();
    for (int o = 128; o > 0; o >>= 1) { if (tid < o) red[tid] += red[tid + o]; __syncthreads(); }
    float rstd = rsqrtf(red[0] / (float)C + 1e-6f);
    int i = 0;
    for (int c = tid; c < C; c += 256) { yr[c] = (v[i] - mu) * rstd * g[c] + b[c]; ++i; }
}

// ---------------------------------------------------------------------------
// Build orthonormal DFT matrices for chunk of side S (W' = S/2+1):
//   F (W'xS): forward real rfft row transform (Fr=cos/sqrtS, Fi=-sin/sqrtS)
//   G (SxS) : forward complex column DFT (e^{-i}), H (SxS): inverse (e^{+i})
//   R (SxW'): Hermitian-weighted inverse row transform (alpha=1 at DC/Nyquist)
// Angles reduced mod S before sin/cos for accuracy.
// ---------------------------------------------------------------------------
__global__ void build_dft_kernel(float* Fr, float* Fi, float* Gr, float* Gi,
                                 float* Hr, float* Hi, float* Rr, float* Ri, int S)
{
    const int Wp = S / 2 + 1;
    const float inv = rsqrtf((float)S);
    const float twoPi = 6.28318530717958647692f;
    const int tid = blockIdx.x * blockDim.x + threadIdx.x;
    const int stride = gridDim.x * blockDim.x;

    for (int i = tid; i < Wp * S; i += stride) {
        int wp = i / S, w = i - wp * S;
        int tt = (wp * w) % S;
        float th = twoPi * (float)tt / (float)S;
        Fr[i] = cosf(th) * inv;
        Fi[i] = -sinf(th) * inv;
    }
    for (int i = tid; i < S * S; i += stride) {
        int u = i / S, h = i - u * S;
        int tt = (u * h) % S;
        float th = twoPi * (float)tt / (float)S;
        float c = cosf(th) * inv, sn = sinf(th) * inv;
        Gr[i] = c;  Gi[i] = -sn;
        Hr[i] = c;  Hi[i] = sn;
    }
    for (int i = tid; i < S * Wp; i += stride) {
        int w = i / Wp, wp = i - w * Wp;
        int tt = (w * wp) % S;
        float th = twoPi * (float)tt / (float)S;
        float a = (wp == 0 || 2 * wp == S) ? 1.0f : 2.0f;
        Rr[i] = a * cosf(th) * inv;
        Ri[i] = -a * sinf(th) * inv;
    }
}

// Split complex filter w_c (C,C,2) into real/imag planes (C,C)
__global__ void deinterleave_kernel(const float* __restrict__ wc,
                                    float* __restrict__ wa, float* __restrict__ wb, int n)
{
    int i = blockIdx.x * blockDim.x + threadIdx.x;
    if (i < n) { wa[i] = wc[2 * i]; wb[i] = wc[2 * i + 1]; }
}

// ---------------------------------------------------------------------------
// Host side
// ---------------------------------------------------------------------------
static void launch_gemm(hipStream_t stream,
                        const float* A1, long long lda1,
                        const float* B1, long long ldb1, long long bsB1o, long long bsB1i,
                        const float* A2, long long lda2,
                        const float* B2, long long ldb2, long long bsB2o, long long bsB2i,
                        float scale2,
                        float* D, long long ldd, long long bsDo, long long bsDi,
                        const float* bias, const float* resid, long long ldr,
                        int M, int N, int K, int batch, int nbi, int act)
{
    dim3 grid((N + BN - 1) / BN, (M + BM - 1) / BM, batch);
    gemm_bf16x3_kernel<<<grid, dim3(128), 0, stream>>>(
        A1, lda1, B1, ldb1, bsB1o, bsB1i,
        A2, lda2, B2, ldb2, bsB2o, bsB2i,
        scale2, D, ldd, bsDo, bsDi, bias, resid, ldr, M, N, K, nbi, act);
}

extern "C" void kernel_launch(void* const* d_in, const int* in_sizes, int n_in,
                              void* d_out, int out_size, void* d_ws, size_t ws_size,
                              hipStream_t stream)
{
    (void)in_sizes; (void)n_in; (void)out_size; (void)ws_size;

    const float* x     = (const float*)d_in[0];
    const float* ln1_g = (const float*)d_in[1];
    const float* ln1_b = (const float*)d_in[2];
    const float* w_c   = (const float*)d_in[3];
    const float* ln2_g = (const float*)d_in[4];
    const float* ln2_b = (const float*)d_in[5];
    const float* w1    = (const float*)d_in[6];
    const float* b1    = (const float*)d_in[7];
    const float* w2    = (const float*)d_in[8];
    const float* b2    = (const float*)d_in[9];
    float* out = (float*)d_out;

    const int Bn = 16, Stot = 4116, C = 768, Fdim = 3072;
    const int rows = Bn * Stot;                 // 65856 tokens
    const int Ss[3]     = {56, 28, 14};
    const int seqOff[3] = {0, 3136, 3920};

    // ---- workspace carve-up (floats) ----
    float* ws = (float*)d_ws;
    size_t off = 0;
    auto take = [&](size_t n) { float* p = ws + off; off += (n + 63) & ~(size_t)63; return p; };

    float* h  = take((size_t)rows * C);         // LN1 out; mixed result in place
    float* y0 = take((size_t)rows * C);         // LN2 out
    const size_t zmax = (size_t)Bn * 56 * 29 * C;
    float* Zr = take(zmax);
    float* Zi = take(zmax);
    float* Yr = take(zmax);
    float* Yi = take(zmax);
    const int SLAB = 8232;                      // 65856 / 8
    float* y1 = take((size_t)SLAB * Fdim);      // MLP hidden slab
    float* wA = take((size_t)C * C);
    float* wB = take((size_t)C * C);

    float *Fr[3], *Fi[3], *Gr[3], *Gi[3], *Hr[3], *Hi[3], *Rr[3], *Ri[3];
    for (int c = 0; c < 3; ++c) {
        int S = Ss[c], Wp = S / 2 + 1;
        Fr[c] = take((size_t)Wp * S); Fi[c] = take((size_t)Wp * S);
        Gr[c] = take((size_t)S * S);  Gi[c] = take((size_t)S * S);
        Hr[c] = take((size_t)S * S);  Hi[c] = take((size_t)S * S);
        Rr[c] = take((size_t)S * Wp); Ri[c] = take((size_t)S * Wp);
    }

    // ---- constants / tables ----
    for (int c = 0; c < 3; ++c)
        build_dft_kernel<<<32, 256, 0, stream>>>(Fr[c], Fi[c], Gr[c], Gi[c],
                                                 Hr[c], Hi[c], Rr[c], Ri[c], Ss[c]);
    {
        int n = C * C;
        deinterleave_kernel<<<(n + 255) / 256, 256, 0, stream>>>(w_c, wA, wB, n);
    }

    // ---- LN1 ----
    layernorm_kernel<<<rows, 256, 0, stream>>>(x, ln1_g, ln1_b, h, C);

    // ---- spectral mixing per chunk ----
    for (int c = 0; c < 3; ++c) {
        const int S = Ss[c], Wp = S / 2 + 1;
        float* hbase = h + (size_t)seqOff[c] * C;
        const long long hOuter = (long long)Stot * C;   // per batch image
        const long long hInner = (long long)S * C;      // per spatial row
        const long long zOuter = (long long)S * Wp * C; // per (b) in freq buf
        const long long zInner = (long long)Wp * C;     // per row in freq buf

        // P1: row rfft (contract W). batch = (b, h-row)
        launch_gemm(stream, Fr[c], S, hbase, C, hOuter, hInner,
                    nullptr, 0, nullptr, 0, 0, 0, 0.0f,
                    Zr, C, zOuter, zInner, nullptr, nullptr, 0,
                    Wp, C, S, Bn * S, S, 0);
        launch_gemm(stream, Fi[c], S, hbase, C, hOuter, hInner,
                    nullptr, 0, nullptr, 0, 0, 0, 0.0f,
                    Zi, C, zOuter, zInner, nullptr, nullptr, 0,
                    Wp, C, S, Bn * S, S, 0);

        // P2: forward column DFT (contract H, complex). batch = b
        launch_gemm(stream, Gr[c], S, Zr, zInner, zOuter, 0,
                    Gi[c], S, Zi, zInner, zOuter, 0, -1.0f,
                    Yr, zInner, zOuter, 0, nullptr, nullptr, 0,
                    S, Wp * C, S, Bn, 1, 0);
        launch_gemm(stream, Gi[c], S, Zr, zInner, zOuter, 0,
                    Gr[c], S, Zi, zInner, zOuter, 0, 1.0f,
                    Yi, zInner, zOuter, 0, nullptr, nullptr, 0,
                    S, Wp * C, S, Bn, 1, 0);

        // P3: complex channel mix (contract C), one big GEMM per component
        const int P = Bn * S * Wp;
        launch_gemm(stream, Yr, C, wA, C, 0, 0,
                    Yi, C, wB, C, 0, 0, -1.0f,
                    Zr, C, 0, 0, nullptr, nullptr, 0,
                    P, C, C, 1, 1, 0);
        launch_gemm(stream, Yr, C, wB, C, 0, 0,
                    Yi, C, wA, C, 0, 0, 1.0f,
                    Zi, C, 0, 0, nullptr, nullptr, 0,
                    P, C, C, 1, 1, 0);

        // P4: inverse column DFT (complex)
        launch_gemm(stream, Hr[c], S, Zr, zInner, zOuter, 0,
                    Hi[c], S, Zi, zInner, zOuter, 0, -1.0f,
                    Yr, zInner, zOuter, 0, nullptr, nullptr, 0,
                    S, Wp * C, S, Bn, 1, 0);
        launch_gemm(stream, Hi[c], S, Zr, zInner, zOuter, 0,
                    Hr[c], S, Zi, zInner, zOuter, 0, 1.0f,
                    Yi, zInner, zOuter, 0, nullptr, nullptr, 0,
                    S, Wp * C, S, Bn, 1, 0);

        // P5: inverse row transform (contract W'), real out, in place into h
        launch_gemm(stream, Rr[c], Wp, Yr, C, zOuter, zInner,
                    Ri[c], Wp, Yi, C, zOuter, zInner, 1.0f,
                    hbase, C, hOuter, hInner, nullptr, nullptr, 0,
                    S, C, Wp, Bn * S, S, 0);
    }

    // ---- LN2 ----
    layernorm_kernel<<<rows, 256, 0, stream>>>(h, ln2_g, ln2_b, y0, C);

    // ---- MLP in token slabs: GELU(y0@W1+b1)@W2 + b2 + x ----
    for (int r0 = 0; r0 < rows; r0 += SLAB) {
        int rs = rows - r0; if (rs > SLAB) rs = SLAB;
        launch_gemm(stream, y0 + (size_t)r0 * C, C, w1, Fdim, 0, 0,
                    nullptr, 0, nullptr, 0, 0, 0, 0.0f,
                    y1, Fdim, 0, 0, b1, nullptr, 0,
                    rs, Fdim, C, 1, 1, 1);
        launch_gemm(stream, y1, Fdim, w2, C, 0, 0,
                    nullptr, 0, nullptr, 0, 0, 0, 0.0f,
                    out + (size_t)r0 * C, C, 0, 0, b2, x + (size_t)r0 * C, C,
                    rs, C, Fdim, 1, 1, 0);
    }
}
